// ApproximateEuclideanAttention_59485297050336
// MI455X (gfx1250) — matrix operand order repaired and checked by reading, not compiled, not run on hardware
//
#include <hip/hip_runtime.h>
#include <hip/hip_bf16.h>

#define BATCHN 2
#define SEQ    8192
#define EMBED  1024
#define HEADS  16
#define HDIM   64
#define LMK    64
#define ROWS   (BATCHN * SEQ)   /* 16384 */
#define BH     (BATCHN * HEADS) /* 32    */

typedef __attribute__((ext_vector_type(16))) __bf16 v16bf;
typedef __attribute__((ext_vector_type(8)))  float  v8f;

union Frag { v16bf v; uint4 q[2]; unsigned int u[8]; };

__device__ __forceinline__ unsigned short f2bf(float f) {
  unsigned int x = __float_as_uint(f);
  x += 0x7fffu + ((x >> 16) & 1u);           // round-to-nearest-even
  return (unsigned short)(x >> 16);
}
__device__ __forceinline__ float bf2f(unsigned int s16) {
  return __uint_as_float((s16 & 0xffffu) << 16);
}

// A fragment (K contiguous: lanes 0-15 K=0..15, lanes 16-31 K=16..31): per lane,
//   dwords base + k0/2 + half*8 + {0..7} (two aligned 128-bit chunks)
__device__ __forceinline__ void loadA(Frag& f, const unsigned int* rowBase,
                                      int k0, int half) {
  const unsigned int* p = rowBase + (k0 >> 1) + half * 8;
  f.q[0] = *(const uint4*)(p);
  f.q[1] = *(const uint4*)(p + 4);
}
// B fragment (K contiguous: lanes 0-15 K=0..15, lanes 16-31 K=16..31): per lane,
// dwords base + k0/2 + half*8 + {0..7} (two aligned 128-bit chunks)
__device__ __forceinline__ void loadB(Frag& f, const unsigned int* colBase,
                                      int k0, int half) {
  const unsigned int* p = colBase + (k0 >> 1) + half * 8;
  f.q[0] = *(const uint4*)(p);
  f.q[1] = *(const uint4*)(p + 4);
}

// ---------------------------------------------------------------- fp32->bf16
__global__ void k_cvt_bf16(const float* __restrict__ src,
                           unsigned short* __restrict__ dst, int n) {
  int i = blockIdx.x * blockDim.x + threadIdx.x;
  if (i < n) dst[i] = f2bf(src[i]);
}

// ------------------------------------------------- big GEMM: (Mx1024)*(1024x1024)^T
// 2x2 register blocking: each wave -> 32x32 output. Workgroup: 128 rows x 64 cols.
// mode 0: Q -> Qb[b,h,n,d]   mode 1: K -> Kb[b,h,n,d]
// mode 2: V -> Vt[b,h,d,n]   mode 3: fp32 out[b,n,e]
__global__ void __launch_bounds__(256) k_gemm1024(
    const unsigned short* __restrict__ A,
    const unsigned short* __restrict__ W0, const unsigned short* __restrict__ W1,
    const unsigned short* __restrict__ W2, const unsigned short* __restrict__ W3,
    const float* __restrict__ b0, const float* __restrict__ b1,
    const float* __restrict__ b2, const float* __restrict__ b3,
    unsigned short* __restrict__ Qb, unsigned short* __restrict__ Kb,
    unsigned short* __restrict__ Vt, float* __restrict__ Of, int mode_base) {
  const int mode = mode_base + (int)blockIdx.z;
  const unsigned short* W; const float* bias;
  if      (mode == 0) { W = W0; bias = b0; }
  else if (mode == 1) { W = W1; bias = b1; }
  else if (mode == 2) { W = W2; bias = b2; }
  else                { W = W3; bias = b3; }

  const int wid = threadIdx.x >> 5, lane = threadIdx.x & 31;
  const int m = lane & 15, half = lane >> 4;
  const int row0 = ((int)blockIdx.x * 4 + (wid >> 1)) * 32;
  const int col0 = ((int)blockIdx.y * 2 + (wid & 1)) * 32;

  const unsigned int* Ap = (const unsigned int*)A;
  const unsigned int* Wp = (const unsigned int*)W;
  const unsigned int* a0p = Ap + (row0 + m) * (EMBED / 2);
  const unsigned int* a1p = Ap + (row0 + 16 + m) * (EMBED / 2);
  const int e0 = col0 + m, e1 = e0 + 16;
  const unsigned int* b0p = Wp + e0 * (EMBED / 2);
  const unsigned int* b1p = Wp + e1 * (EMBED / 2);

  v8f c00 = {0.f,0.f,0.f,0.f,0.f,0.f,0.f,0.f};
  v8f c01 = c00, c10 = c00, c11 = c00;
  for (int k0 = 0; k0 < EMBED; k0 += 32) {
    int kp = k0 + 64; kp = (kp < EMBED) ? kp : (EMBED - 32);   // branch-free
    __builtin_prefetch(a0p + (kp >> 1), 0, 3);                 // global_prefetch_b8
    __builtin_prefetch(a1p + (kp >> 1), 0, 3);
    Frag fa0, fa1, fb0, fb1;
    loadA(fa0, a0p, k0, half);
    loadA(fa1, a1p, k0, half);
    loadB(fb0, b0p, k0, half);
    loadB(fb1, b1p, k0, half);
    c00 = __builtin_amdgcn_wmma_f32_16x16x32_bf16(false, fa0.v, false, fb0.v, (short)0, c00, false, false);
    c01 = __builtin_amdgcn_wmma_f32_16x16x32_bf16(false, fa0.v, false, fb1.v, (short)0, c01, false, false);
    c10 = __builtin_amdgcn_wmma_f32_16x16x32_bf16(false, fa1.v, false, fb0.v, (short)0, c10, false, false);
    c11 = __builtin_amdgcn_wmma_f32_16x16x32_bf16(false, fa1.v, false, fb1.v, (short)0, c11, false, false);
  }

  const float bv0 = bias[e0], bv1 = bias[e1];
  if (mode == 3) {
#pragma unroll
    for (int v = 0; v < 8; ++v) {
      const int r0 = row0 + half * 8 + v, r1 = r0 + 16;
      Of[r0 * EMBED + e0] = c00[v] + bv0;
      Of[r0 * EMBED + e1] = c01[v] + bv1;
      Of[r1 * EMBED + e0] = c10[v] + bv0;
      Of[r1 * EMBED + e1] = c11[v] + bv1;
    }
  } else if (mode == 2) {
    auto stV = [&](int r, int e, float val) {
      const int b = r >> 13, n = r & (SEQ - 1), h = e >> 6, d = e & 63;
      Vt[(((b * HEADS + h) << 6) + d) * SEQ + n] = f2bf(val);
    };
#pragma unroll
    for (int v = 0; v < 8; ++v) {
      const int r0 = row0 + half * 8 + v, r1 = r0 + 16;
      stV(r0, e0, c00[v] + bv0); stV(r0, e1, c01[v] + bv1);
      stV(r1, e0, c10[v] + bv0); stV(r1, e1, c11[v] + bv1);
    }
  } else {
    unsigned short* __restrict__ P = (mode == 0) ? Qb : Kb;
    auto stQK = [&](int r, int e, float val) {
      const int b = r >> 13, n = r & (SEQ - 1), h = e >> 6, d = e & 63;
      P[((((b * HEADS + h) << 13) + n) << 6) + d] = f2bf(val);
    };
#pragma unroll
    for (int v = 0; v < 8; ++v) {
      const int r0 = row0 + half * 8 + v, r1 = r0 + 16;
      stQK(r0, e0, c00[v] + bv0); stQK(r0, e1, c01[v] + bv1);
      stQK(r1, e0, c10[v] + bv0); stQK(r1, e1, c11[v] + bv1);
    }
  }
}

// ---------------------------------------------------------------- row norms
__global__ void k_rownorm(const unsigned short* __restrict__ X,
                          float* __restrict__ out, int nrows) {
  int r = blockIdx.x * blockDim.x + threadIdx.x;
  if (r >= nrows) return;
  const unsigned int* p = (const unsigned int*)X + r * 32;
  float s = 0.f;
#pragma unroll
  for (int i = 0; i < 32; ++i) {
    unsigned int u = p[i];
    float a = bf2f(u), b = bf2f(u >> 16);
    s += a * a + b * b;
  }
  out[r] = s;
}

// ------------- Phi = exp(-max(|x|^2+|l|^2-2 x.l,0)/tau), WMMA for x.l
// Block covers 64 n-rows; wave: 2 row-tiles sharing one landmark B-fragment.
__global__ void __launch_bounds__(256) k_phi(
    const unsigned short* __restrict__ Xb, const float* __restrict__ xn,
    const unsigned short* __restrict__ Kb, const float* __restrict__ kn,
    const int* __restrict__ lidx, const float* __restrict__ tempP,
    unsigned short* __restrict__ PhiQ, unsigned short* __restrict__ PhiKT,
    float* __restrict__ PKpart, int isK) {
  __shared__ float spart[4][64];
  const int bh = blockIdx.y;
  const int n0 = (int)blockIdx.x * 64;
  const int wid = threadIdx.x >> 5, lane = threadIdx.x & 31;
  const int m = lane & 15, half = lane >> 4;
  const int rg = wid >> 2, ct = wid & 3;
  const int nrow0 = n0 + rg * 32;
  const int l = ct * 16 + m;
  const int lrow = lidx[l];
  const float temp = *tempP;

  const unsigned int* Ap0 = (const unsigned int*)Xb + (bh * SEQ + nrow0 + m) * 32;
  const unsigned int* Ap1 = Ap0 + 16 * 32;
  const unsigned int* Bp  = (const unsigned int*)Kb + (bh * SEQ + lrow) * 32;

  v8f c0 = {0.f,0.f,0.f,0.f,0.f,0.f,0.f,0.f};
  v8f c1 = c0;
#pragma unroll
  for (int k0 = 0; k0 < 64; k0 += 32) {
    Frag fa0, fa1, fb;
    loadA(fa0, Ap0, k0, half);
    loadA(fa1, Ap1, k0, half);
    loadB(fb, Bp, k0, half);
    c0 = __builtin_amdgcn_wmma_f32_16x16x32_bf16(false, fa0.v, false, fb.v, (short)0, c0, false, false);
    c1 = __builtin_amdgcn_wmma_f32_16x16x32_bf16(false, fa1.v, false, fb.v, (short)0, c1, false, false);
  }

  const float lnv = kn[bh * SEQ + lrow];
  float lsum = 0.f;
#pragma unroll
  for (int t = 0; t < 2; ++t) {
    const v8f& c = (t == 0) ? c0 : c1;
#pragma unroll
    for (int v = 0; v < 8; ++v) {
      const int n = nrow0 + t * 16 + half * 8 + v;
      const float d2 = fmaxf(xn[bh * SEQ + n] + lnv - 2.f * c[v], 0.f);
      const float ph = __expf(-d2 / temp);
      if (!isK) {
        PhiQ[((bh * SEQ + n) << 6) + l] = f2bf(ph);
      } else {
        PhiKT[(bh * 64 + l) * SEQ + n] = f2bf(ph);
        lsum += ph;
      }
    }
  }
  if (isK) {
    spart[rg * 2 + half][l] = lsum;     // unique (rg,half,ct,m) per slot
    __syncthreads();
    if (threadIdx.x < 64) {
      float s = spart[0][threadIdx.x] + spart[1][threadIdx.x] +
                spart[2][threadIdx.x] + spart[3][threadIdx.x];
      PKpart[(bh * 128 + (int)blockIdx.x) * 64 + threadIdx.x] = s;
    }
  }
}

__global__ void k_reduce_pk(const float* __restrict__ part,
                            float* __restrict__ PhiKsum) {
  int i = blockIdx.x * blockDim.x + threadIdx.x;
  if (i >= BH * 64) return;
  int bh = i >> 6, l = i & 63;
  float s = 0.f;
  for (int b = 0; b < 128; ++b) s += part[(bh * 128 + b) * 64 + l];
  PhiKsum[i] = s;
}

// ------------------- Z = PhiK^T * V  (split-K over n, fixed-order reduce)
// Wave: 2 l-tiles sharing one V B-fragment.
__global__ void __launch_bounds__(256) k_z(
    const unsigned short* __restrict__ PhiKT, const unsigned short* __restrict__ Vt,
    float* __restrict__ Zpart) {
  const int bh = blockIdx.y;
  const int ks = blockIdx.x;
  const int n0 = ks * 1024;
  const int wid = threadIdx.x >> 5, lane = threadIdx.x & 31;
  const int m = lane & 15, half = lane >> 4;
  const int lg = wid >> 2, dt = wid & 3;
  const int lt0 = lg * 2;

  const unsigned int* Ap0 = (const unsigned int*)PhiKT + (bh * 64 + lt0 * 16 + m) * (SEQ / 2) + (n0 >> 1);
  const unsigned int* Ap1 = Ap0 + 16 * (SEQ / 2);
  const unsigned int* Bp  = (const unsigned int*)Vt + (bh * 64 + dt * 16 + m) * (SEQ / 2) + (n0 >> 1);

  v8f c0 = {0.f,0.f,0.f,0.f,0.f,0.f,0.f,0.f};
  v8f c1 = c0;
  for (int k0 = 0; k0 < 1024; k0 += 32) {
    Frag fa0, fa1, fb;
    loadA(fa0, Ap0, k0, half);
    loadA(fa1, Ap1, k0, half);
    loadB(fb, Bp, k0, half);
    c0 = __builtin_amdgcn_wmma_f32_16x16x32_bf16(false, fa0.v, false, fb.v, (short)0, c0, false, false);
    c1 = __builtin_amdgcn_wmma_f32_16x16x32_bf16(false, fa1.v, false, fb.v, (short)0, c1, false, false);
  }
#pragma unroll
  for (int t = 0; t < 2; ++t) {
    const v8f& c = (t == 0) ? c0 : c1;
#pragma unroll
    for (int v = 0; v < 8; ++v) {
      const int l = (lt0 + t) * 16 + half * 8 + v;
      const int d = dt * 16 + m;
      Zpart[(ks * BH + bh) * 4096 + l * 64 + d] = c[v];
    }
  }
}

__global__ void k_reduce_z(const float* __restrict__ part, float* __restrict__ Z) {
  int i = blockIdx.x * blockDim.x + threadIdx.x;
  if (i >= BH * 4096) return;
  int bh = i >> 12, rem = i & 4095;
  float s = 0.f;
  for (int ks = 0; ks < 8; ++ks) s += part[(ks * BH + bh) * 4096 + rem];
  Z[i] = s;
}

// -------- per-(b,h): build W kernel, Gauss-Jordan solve [W | Z | sum] in LDS
__global__ void __launch_bounds__(64) k_solve(
    const unsigned short* __restrict__ Kb, const float* __restrict__ kn,
    const int* __restrict__ lidx, const float* __restrict__ tempP,
    const float* __restrict__ Z, const float* __restrict__ PhiKsum,
    unsigned short* __restrict__ Ybf, float* __restrict__ wv) {
  __shared__ float As[64][129];
  __shared__ float Lm[64][65];
  __shared__ float lnv[64];
  const int r = threadIdx.x, bh = blockIdx.x;
  const int lrow = lidx[r];
  const unsigned int* Kp = (const unsigned int*)Kb + (bh * SEQ + lrow) * 32;
#pragma unroll
  for (int i = 0; i < 32; ++i) {
    unsigned int u = Kp[i];
    Lm[r][2 * i]     = bf2f(u);
    Lm[r][2 * i + 1] = bf2f(u >> 16);
  }
  lnv[r] = kn[bh * SEQ + lrow];
  __syncthreads();
  const float temp = *tempP;
  for (int j = 0; j < 64; ++j) {
    float dot = 0.f;
#pragma unroll 8
    for (int d = 0; d < 64; ++d) dot += Lm[r][d] * Lm[j][d];
    const float d2 = fmaxf(lnv[r] + lnv[j] - 2.f * dot, 0.f);
    As[r][j] = __expf(-d2 / temp) + ((r == j) ? 1e-6f : 0.f);
  }
  for (int j = 0; j < 64; ++j) As[r][64 + j] = Z[bh * 4096 + r * 64 + j];
  As[r][128] = PhiKsum[bh * 64 + r];
  __syncthreads();
  for (int cc = 0; cc < 64; ++cc) {           // Gauss-Jordan, SPD -> no pivoting
    if (r != cc) {
      const float f = As[r][cc] / As[cc][cc];
      for (int j = cc; j <= 128; ++j) As[r][j] -= f * As[cc][j];
    }
    __syncthreads();
  }
  const float dinv = 1.0f / As[r][r];
  for (int j = 0; j < 64; ++j)
    Ybf[(bh * 64 + j) * 64 + r] = f2bf(As[r][64 + j] * dinv);  // Y^T[d][l]
  wv[bh * 64 + r] = As[r][128] * dinv;
}

// ---------------- out = (PhiQ @ Y) / max(PhiQ @ w, 1e-10) -> attn[b,n,h*64+d]
// Block covers 64 n-rows; wave: 2 row-tiles sharing one Y B-fragment.
__global__ void __launch_bounds__(256) k_out(
    const unsigned short* __restrict__ PhiQ, const unsigned short* __restrict__ Ybf,
    const float* __restrict__ wv, unsigned short* __restrict__ attn) {
  __shared__ float snorm[64];
  const int bh = blockIdx.y, n0 = (int)blockIdx.x * 64;
  const int tid = threadIdx.x;
  if (tid < 64) {
    const unsigned int* p = (const unsigned int*)PhiQ + (bh * SEQ + n0 + tid) * 32;
    const float* wp = wv + bh * 64;
    float s = 0.f;
#pragma unroll
    for (int i = 0; i < 32; ++i) {
      unsigned int u = p[i];
      s += bf2f(u) * wp[2 * i] + bf2f(u >> 16) * wp[2 * i + 1];
    }
    snorm[tid] = fmaxf(s, 1e-10f);
  }
  __syncthreads();
  const int wid = tid >> 5, lane = tid & 31;
  const int m = lane & 15, half = lane >> 4;
  const int rg = wid >> 2, ct = wid & 3;
  const int nrow0 = n0 + rg * 32, d0 = ct * 16;
  const unsigned int* Ap0 = (const unsigned int*)PhiQ + (bh * SEQ + nrow0 + m) * 32;
  const unsigned int* Ap1 = Ap0 + 16 * 32;
  const unsigned int* Bp  = (const unsigned int*)Ybf + (bh * 64 + d0 + m) * 32;
  v8f c0 = {0.f,0.f,0.f,0.f,0.f,0.f,0.f,0.f};
  v8f c1 = c0;
#pragma unroll
  for (int k0 = 0; k0 < 64; k0 += 32) {
    Frag fa0, fa1, fb;
    loadA(fa0, Ap0, k0, half);
    loadA(fa1, Ap1, k0, half);
    loadB(fb, Bp, k0, half);
    c0 = __builtin_amdgcn_wmma_f32_16x16x32_bf16(false, fa0.v, false, fb.v, (short)0, c0, false, false);
    c1 = __builtin_amdgcn_wmma_f32_16x16x32_bf16(false, fa1.v, false, fb.v, (short)0, c1, false, false);
  }
  const int b = bh >> 4, h = bh & 15;
#pragma unroll
  for (int t = 0; t < 2; ++t) {
    const v8f& c = (t == 0) ? c0 : c1;
#pragma unroll
    for (int v = 0; v < 8; ++v) {
      const int nl = rg * 32 + t * 16 + half * 8 + v;
      const int n = n0 + nl;
      attn[(((b << 13) + n) * EMBED) + h * 64 + d0 + m] = f2bf(c[v] / snorm[nl]);
    }
  }
}

// ============================================================== launcher
extern "C" void kernel_launch(void* const* d_in, const int* in_sizes, int n_in,
                              void* d_out, int out_size, void* d_ws, size_t ws_size,
                              hipStream_t stream) {
  (void)in_sizes; (void)n_in; (void)out_size; (void)ws_size;
  const float* query = (const float*)d_in[0];
  const float* Wq = (const float*)d_in[1];  const float* bq = (const float*)d_in[2];
  const float* Wk = (const float*)d_in[3];  const float* bk = (const float*)d_in[4];
  const float* Wv = (const float*)d_in[5];  const float* bv = (const float*)d_in[6];
  const float* Wo = (const float*)d_in[7];  const float* bo = (const float*)d_in[8];
  const float* temp = (const float*)d_in[9];
  const int*   lidx = (const int*)d_in[10];
  float* out = (float*)d_out;

  char* ws = (char*)d_ws;
  auto take = [&](size_t bytes) -> char* {
    char* p = ws; ws += (bytes + 255) & ~size_t(255); return p;
  };
  unsigned short* qbf   = (unsigned short*)take((size_t)ROWS * EMBED * 2);
  unsigned short* wqb   = (unsigned short*)take((size_t)EMBED * EMBED * 2);
  unsigned short* wkb   = (unsigned short*)take((size_t)EMBED * EMBED * 2);
  unsigned short* wvb   = (unsigned short*)take((size_t)EMBED * EMBED * 2);
  unsigned short* wob   = (unsigned short*)take((size_t)EMBED * EMBED * 2);
  unsigned short* Qb    = (unsigned short*)take((size_t)BH * SEQ * 64 * 2);
  unsigned short* Kb    = (unsigned short*)take((size_t)BH * SEQ * 64 * 2);
  unsigned short* Vt    = (unsigned short*)take((size_t)BH * SEQ * 64 * 2);
  float*          qn    = (float*)take((size_t)BH * SEQ * 4);
  float*          kn    = (float*)take((size_t)BH * SEQ * 4);
  unsigned short* PhiQ  = (unsigned short*)take((size_t)BH * SEQ * 64 * 2);
  unsigned short* PhiKT = (unsigned short*)take((size_t)BH * SEQ * 64 * 2);
  float*          PKp   = (float*)take((size_t)BH * 128 * 64 * 4);
  float*          PKsum = (float*)take((size_t)BH * 64 * 4);
  float*          Zpart = (float*)take((size_t)8 * BH * 4096 * 4);
  float*          Zb    = (float*)take((size_t)BH * 4096 * 4);
  unsigned short* Ybf   = (unsigned short*)take((size_t)BH * 64 * 64 * 2);
  float*          wvv   = (float*)take((size_t)BH * 64 * 4);
  unsigned short* attn  = (unsigned short*)take((size_t)ROWS * EMBED * 2);

  const int nQ = ROWS * EMBED, nW = EMBED * EMBED;
  k_cvt_bf16<<<(nQ + 255) / 256, 256, 0, stream>>>(query, qbf, nQ);
  k_cvt_bf16<<<(nW + 255) / 256, 256, 0, stream>>>(Wq, wqb, nW);
  k_cvt_bf16<<<(nW + 255) / 256, 256, 0, stream>>>(Wk, wkb, nW);
  k_cvt_bf16<<<(nW + 255) / 256, 256, 0, stream>>>(Wv, wvb, nW);
  k_cvt_bf16<<<(nW + 255) / 256, 256, 0, stream>>>(Wo, wob, nW);

  dim3 gQKV(128, 16, 3);
  k_gemm1024<<<gQKV, 256, 0, stream>>>(qbf, wqb, wkb, wvb, wob,
                                       bq, bk, bv, bo, Qb, Kb, Vt, out, 0);

  k_rownorm<<<(BH * SEQ + 255) / 256, 256, 0, stream>>>(Qb, qn, BH * SEQ);
  k_rownorm<<<(BH * SEQ + 255) / 256, 256, 0, stream>>>(Kb, kn, BH * SEQ);

  dim3 gP(128, 32, 1);
  k_phi<<<gP, 256, 0, stream>>>(Qb, qn, Kb, kn, lidx, temp, PhiQ, PhiKT, PKp, 0);
  k_phi<<<gP, 256, 0, stream>>>(Kb, kn, Kb, kn, lidx, temp, PhiQ, PhiKT, PKp, 1);
  k_reduce_pk<<<(BH * 64 + 255) / 256, 256, 0, stream>>>(PKp, PKsum);

  dim3 gZ(8, 32, 1);
  k_z<<<gZ, 256, 0, stream>>>(PhiKT, Vt, Zpart);
  k_reduce_z<<<(BH * 4096 + 255) / 256, 256, 0, stream>>>(Zpart, Zb);

  k_solve<<<32, 64, 0, stream>>>(Kb, kn, lidx, temp, Zb, PKsum, Ybf, wvv);

  k_out<<<gP, 256, 0, stream>>>(PhiQ, Ybf, wvv, attn);

  dim3 gO(128, 16, 1);
  k_gemm1024<<<gO, 256, 0, stream>>>(attn, wqb, wkb, wvb, wob,
                                     bq, bk, bv, bo, Qb, Kb, Vt, out, 3);
}
